// NeighborVoxelSAModuleMSGV2_56719338111373
// MI455X (gfx1250) — compile-verified, matrix operand order checked by reading, not run on hardware
//
#include <hip/hip_runtime.h>
#include <hip/hip_bf16.h>

// ---------------- problem constants (from reference) ----------------
#define EPS      1e-5f
#define CIN      64
#define CMID     32
#define COUT     64
#define N_PTS    16384
#define M_GRID   65536
#define BB       2
#define ZD       11
#define YD       200
#define XD       176

// workspace layout (float offsets)
#define OFF_FPAD   0ull                                   // 2 * (N_PTS+1) * CMID
#define SZ_FPAD    (2ull * (N_PTS + 1) * CMID)
#define OFF_PF     (OFF_FPAD + SZ_FPAD)                   // 2 * 64 * CMID
#define SZ_PF      (2ull * 64 * CMID)
#define OFF_NF     (OFF_PF + SZ_PF)                       // 2 * M_GRID * CMID
#define SZ_NF      (2ull * M_GRID * CMID)
#define OFF_TMPD   (OFF_NF + SZ_NF)                       // 2 * M_GRID * COUT
#define SZ_TMPD    (2ull * M_GRID * COUT)
#define OFF_STATSA (OFF_TMPD + SZ_TMPD)                   // per k: mean[32] var[32]
#define SZ_STATSA  (2ull * 64)
#define OFF_STATSD (OFF_STATSA + SZ_STATSA)               // per k: mean[64] var[64]

typedef __attribute__((ext_vector_type(16))) _Float16 v16h;
typedef __attribute__((ext_vector_type(8)))  float    v8f;
typedef __attribute__((ext_vector_type(4)))  int      v4i;

#define AS_GLOBAL __attribute__((address_space(1)))
#define AS_LDS    __attribute__((address_space(3)))

#if __has_builtin(__builtin_amdgcn_global_load_async_to_lds_b128)
#define HAVE_ASYNC_LDS 1
#endif

__device__ __forceinline__ void wait_async0() {
#if __has_builtin(__builtin_amdgcn_s_wait_asynccnt)
    __builtin_amdgcn_s_wait_asynccnt(0);
#else
    asm volatile("s_wait_asynccnt 0" ::: "memory");
#endif
}

// Load one 16-lane-row fragment (16 halves per lane) from f32 row-major data.
// ISA 16-bit A layout: lanes 0-15 hold K = 0..7 and 16..23; lanes 16-31 hold
// K = 8..15 and 24..31 -> two contiguous 8-float runs per lane, base 0 or 8.
__device__ __forceinline__ v16h load_frag_f32(const float* __restrict__ rowptr, int lane) {
    const int base = (lane & 16) ? 8 : 0;
    const float4 q0 = *(const float4*)(rowptr + base);
    const float4 q1 = *(const float4*)(rowptr + base + 4);
    const float4 q2 = *(const float4*)(rowptr + base + 16);
    const float4 q3 = *(const float4*)(rowptr + base + 20);
    v16h f;
    f[0]  = (_Float16)q0.x; f[1]  = (_Float16)q0.y; f[2]  = (_Float16)q0.z; f[3]  = (_Float16)q0.w;
    f[4]  = (_Float16)q1.x; f[5]  = (_Float16)q1.y; f[6]  = (_Float16)q1.z; f[7]  = (_Float16)q1.w;
    f[8]  = (_Float16)q2.x; f[9]  = (_Float16)q2.y; f[10] = (_Float16)q2.z; f[11] = (_Float16)q2.w;
    f[12] = (_Float16)q3.x; f[13] = (_Float16)q3.y; f[14] = (_Float16)q3.z; f[15] = (_Float16)q3.w;
    return f;
}

// D[M,N] = A[M,K] * W[N,K]^T with N = TN*16 (whole output width per wave).
// Block = 256 threads = 8 waves; each wave owns one 16-row M tile; the W
// matrix (N*K floats, 8KB) is staged once per block into LDS (async if
// available). Fully unrolled -> TN*(K/32) static WMMAs per kernel.
template<int K, int TN>
__global__ void __launch_bounds__(256)
wmma_gemm_lds(const float* __restrict__ A,
              const float* __restrict__ W,
              float* __restrict__ D, int M) {
    constexpr int N = TN * 16;
    __shared__ float ldsW[N * K];

    const int tid = threadIdx.x;
#if HAVE_ASYNC_LDS
    {
        constexpr int n4 = (N * K) / 4;
        const float4* src = (const float4*)W;
        float4* dst = (float4*)ldsW;
        for (int i = tid; i < n4; i += 256) {
            __builtin_amdgcn_global_load_async_to_lds_b128(
                (AS_GLOBAL v4i*)(src + i), (AS_LDS v4i*)(dst + i), 0, 0);
        }
        wait_async0();
    }
#else
    for (int i = tid; i < N * K; i += 256) ldsW[i] = W[i];
#endif
    __syncthreads();

    const int lane  = tid & 31;
    const int wv    = tid >> 5;
    const int tileM = blockIdx.x * 8 + wv;     // M/128 blocks, exact
    const int arow  = tileM * 16 + (lane & 15);

    v8f c[TN] = {};
#pragma unroll
    for (int kb = 0; kb < K; kb += 32) {
        const v16h a = load_frag_f32(A + (size_t)arow * K + kb, lane);
#pragma unroll
        for (int t = 0; t < TN; ++t) {
            const v16h b = load_frag_f32(ldsW + (t * 16 + (lane & 15)) * K + kb, lane);
            c[t] = __builtin_amdgcn_wmma_f32_16x16x32_f16(false, a, false, b,
                                                          (short)0, c[t], false, false);
        }
    }

    // C/D layout: VGPR g -> row M = g (lanes 0-15) or g+8 (lanes 16-31), col = lane&15
    const int drow0 = tileM * 16 + ((lane & 16) ? 8 : 0);
#pragma unroll
    for (int t = 0; t < TN; ++t) {
        const int dcol = t * 16 + (lane & 15);
#pragma unroll
        for (int g = 0; g < 8; ++g)
            D[(size_t)(drow0 + g) * N + dcol] = c[t][g];
    }
}

// Column mean / biased variance over `rows` rows; one block per column.
__global__ void colstats_kernel(const float* __restrict__ src, int rows, int cols,
                                float* __restrict__ mean, float* __restrict__ var) {
    const int c = blockIdx.x;
    __shared__ float ss[256];
    __shared__ float ss2[256];
    float s = 0.f, s2 = 0.f;
    for (int i = threadIdx.x; i < rows; i += blockDim.x) {
        float v = src[(size_t)i * cols + c];
        s += v; s2 += v * v;
    }
    ss[threadIdx.x] = s; ss2[threadIdx.x] = s2;
    __syncthreads();
    for (int off = 128; off > 0; off >>= 1) {
        if ((int)threadIdx.x < off) {
            ss[threadIdx.x]  += ss[threadIdx.x + off];
            ss2[threadIdx.x] += ss2[threadIdx.x + off];
        }
        __syncthreads();
    }
    if (threadIdx.x == 0) {
        float mu = ss[0] / (float)rows;
        mean[c] = mu;
        var[c]  = ss2[0] / (float)rows - mu * mu;
    }
}

// In-place BN on fpad rows 1..N_PTS; zero pad row 0.
__global__ void bn_fpad_kernel(float* __restrict__ fpad_k,
                               const float* __restrict__ mean,
                               const float* __restrict__ var,
                               const float* __restrict__ gamma,
                               const float* __restrict__ beta) {
    const size_t t = (size_t)blockIdx.x * blockDim.x + threadIdx.x;
    const size_t total = (size_t)(N_PTS + 1) * CMID;
    if (t >= total) return;
    const int c = (int)(t & (CMID - 1));
    const size_t i = t >> 5;
    if (i == 0) { fpad_k[t] = 0.f; return; }
    float v = fpad_k[t];
    fpad_k[t] = (v - mean[c]) * rsqrtf(var[c] + EPS) * gamma[c] + beta[c];
}

// pf[k][j][c] = bn(off[j] @ W_pos[k][c].T)  — one wave per k, lane = channel.
__global__ void pos_bn_kernel(const float* __restrict__ W_pos,
                              const float* __restrict__ g_pos,
                              const float* __restrict__ b_pos,
                              float* __restrict__ pf) {
    const int k  = blockIdx.x;
    const int c  = threadIdx.x;          // 0..31
    const int ln = k + 1;                // log2(n): n = 2 (r=1) or 4 (r=2)
    const int n = 1 << ln, msk = n - 1, r = n >> 1;
    const int noff = 1 << (3 * ln);

    const float* Wp = W_pos + (size_t)k * CMID * 3 + (size_t)c * 3;
    const float w0 = Wp[0], w1 = Wp[1], w2 = Wp[2];

    float s = 0.f, s2 = 0.f;
    for (int j = 0; j < noff; ++j) {
        float ox = (float)((j & msk) - r);
        float oy = (float)(((j >> ln) & msk) - r);
        float oz = (float)((j >> (2 * ln)) - r);
        float y = ox * w0 + oy * w1 + oz * w2;
        s += y; s2 += y * y;
    }
    const float mu  = s / (float)noff;
    const float vr  = s2 / (float)noff - mu * mu;
    const float sc  = rsqrtf(vr + EPS) * g_pos[k * CMID + c];
    const float bia = b_pos[k * CMID + c];

    for (int j = 0; j < noff; ++j) {
        float ox = (float)((j & msk) - r);
        float oy = (float)(((j >> ln) & msk) - r);
        float oz = (float)((j >> (2 * ln)) - r);
        float y = ox * w0 + oy * w1 + oz * w2;
        pf[((size_t)k * 64 + j) * CMID + c] = (y - mu) * sc + bia;
    }
}

// One wave per (grid point m, scale k); lane = channel.
// Voxel indices are computed lane-parallel (1-2 per lane) and broadcast with
// __shfl (ds_bpermute on wave32) instead of 8/64 redundant all-lane vm loads.
__global__ void gather_max_kernel(const int* __restrict__ new_coords,
                                  const int* __restrict__ vm,
                                  const float* __restrict__ fpad,
                                  const float* __restrict__ pf,
                                  float* __restrict__ nf) {
    const int lane = threadIdx.x & 31;
    const int wid  = (int)((blockIdx.x * blockDim.x + threadIdx.x) >> 5);
    const int m = wid >> 1;
    const int k = wid & 1;
    if (m >= M_GRID) return;

    const int4 ncv = ((const int4*)new_coords)[m];   // [b, x, y, z]
    const int b = ncv.x, xc = ncv.y, yc = ncv.z, zc = ncv.w;

    const int ln = k + 1;
    const int n = 1 << ln, msk = n - 1, r = n >> 1;
    const int noff = 1 << (3 * ln);                  // 8 or 64 (power of two)

    const float* __restrict__ fp  = fpad + (size_t)k * (N_PTS + 1) * CMID;
    const float* __restrict__ pfk = pf + (size_t)k * 64 * CMID;
    __builtin_prefetch(pfk + lane, 0, 0);

    // reference mapping: zi += off_x (fastest), yi += off_y, xi += off_z
    auto calc_idx = [&](int j) -> int {
        const int o1 = (j & msk) - r;
        const int o2 = ((j >> ln) & msk) - r;
        const int o3 = (j >> (2 * ln)) - r;
        const int zi = min(max(zc + o1, 0), ZD - 1);
        const int yi = min(max(yc + o2, 0), YD - 1);
        const int xi = min(max(xc + o3, 0), XD - 1);
        return vm[((b * ZD + zi) * YD + yi) * XD + xi];
    };
    const int idx0 = calc_idx(lane & (noff - 1));
    const int idx1 = (noff > 32) ? calc_idx((lane + 32) & (noff - 1)) : 0;

    float acc = 0.f;  // relu output >= 0 and noff >= 8
    const int nj0 = noff < 32 ? noff : 32;
    for (int j = 0; j < nj0; ++j) {
        const int idx = __shfl(idx0, j, 32);
        const float v = fp[(size_t)(idx + 1) * CMID + lane] + pfk[j * CMID + lane];
        acc = fmaxf(acc, fmaxf(v, 0.f));
    }
    if (noff > 32) {                                  // wave-uniform
        for (int j = 32; j < noff; ++j) {
            const int idx = __shfl(idx1, j - 32, 32);
            const float v = fp[(size_t)(idx + 1) * CMID + lane] + pfk[j * CMID + lane];
            acc = fmaxf(acc, fmaxf(v, 0.f));
        }
    }
    nf[((size_t)k * M_GRID + m) * CMID + lane] = acc;
}

// out[m][k*COUT + c] = relu(bn(tmpD_k[m][c]))
__global__ void bn_relu_out_kernel(const float* __restrict__ tmpD_k,
                                   const float* __restrict__ mean,
                                   const float* __restrict__ var,
                                   const float* __restrict__ gamma,
                                   const float* __restrict__ beta,
                                   float* __restrict__ out, int k) {
    const size_t t = (size_t)blockIdx.x * blockDim.x + threadIdx.x;
    const size_t total = (size_t)M_GRID * COUT;
    if (t >= total) return;
    const int c = (int)(t & (COUT - 1));
    const size_t m = t >> 6;
    float v = (tmpD_k[t] - mean[c]) * rsqrtf(var[c] + EPS) * gamma[c] + beta[c];
    out[m * (2 * COUT) + (size_t)k * COUT + c] = fmaxf(v, 0.f);
}

extern "C" void kernel_launch(void* const* d_in, const int* in_sizes, int n_in,
                              void* d_out, int out_size, void* d_ws, size_t ws_size,
                              hipStream_t stream) {
    (void)in_sizes; (void)n_in; (void)out_size; (void)ws_size;
    // setup_inputs order:
    // 0 xyz, 1 xyz_batch_cnt, 2 new_xyz, 3 new_xyz_batch_cnt, 4 new_coords,
    // 5 features, 6 voxel2point_indices, 7 W_in, 8 g_in, 9 b_in,
    // 10 W_pos, 11 g_pos, 12 b_pos, 13 W_out, 14 g_out, 15 b_out
    const int*   new_coords = (const int*)d_in[4];
    const float* features   = (const float*)d_in[5];
    const int*   vm         = (const int*)d_in[6];
    const float* W_in       = (const float*)d_in[7];
    const float* g_in       = (const float*)d_in[8];
    const float* b_in       = (const float*)d_in[9];
    const float* W_pos      = (const float*)d_in[10];
    const float* g_pos      = (const float*)d_in[11];
    const float* b_pos      = (const float*)d_in[12];
    const float* W_out      = (const float*)d_in[13];
    const float* g_out      = (const float*)d_in[14];
    const float* b_out      = (const float*)d_in[15];
    float* out = (float*)d_out;
    float* ws  = (float*)d_ws;

    float* fpad   = ws + OFF_FPAD;
    float* pf     = ws + OFF_PF;
    float* nf     = ws + OFF_NF;
    float* tmpD   = ws + OFF_TMPD;
    float* statsA = ws + OFF_STATSA;   // per k: [mean(32) | var(32)]
    float* statsD = ws + OFF_STATSD;   // per k: [mean(64) | var(64)]

    // ---- stage 1: f = features @ W_in[k].T  -> fpad rows 1..N_PTS ----
    for (int k = 0; k < 2; ++k) {
        float* fk = fpad + (size_t)k * (N_PTS + 1) * CMID;
        // 16384 rows / 128 rows-per-block = 128 blocks
        wmma_gemm_lds<CIN, 2><<<128, 256, 0, stream>>>(
            features, W_in + (size_t)k * CMID * CIN, fk + CMID, N_PTS);
        colstats_kernel<<<CMID, 256, 0, stream>>>(
            fk + CMID, N_PTS, CMID, statsA + k * 64, statsA + k * 64 + 32);
        const size_t tot = (size_t)(N_PTS + 1) * CMID;
        bn_fpad_kernel<<<(unsigned)((tot + 255) / 256), 256, 0, stream>>>(
            fk, statsA + k * 64, statsA + k * 64 + 32,
            g_in + k * CMID, b_in + k * CMID);
    }

    // ---- stage 2a: positional features ----
    pos_bn_kernel<<<2, 32, 0, stream>>>(W_pos, g_pos, b_pos, pf);

    // ---- stage 2b: gather + relu + max-pool ----
    // waves = M_GRID*2 = 131072 -> 16384 blocks of 8 waves
    gather_max_kernel<<<16384, 256, 0, stream>>>(new_coords, vm, fpad, pf, nf);

    // ---- stage 3: out = relu(bn(nf @ W_out[k].T)) ----
    for (int k = 0; k < 2; ++k) {
        const float* nfk = nf + (size_t)k * M_GRID * CMID;
        float* tk = tmpD + (size_t)k * M_GRID * COUT;
        // 65536 rows / 128 rows-per-block = 512 blocks
        wmma_gemm_lds<CMID, 4><<<512, 256, 0, stream>>>(
            nfk, W_out + (size_t)k * COUT * CMID, tk, M_GRID);
        colstats_kernel<<<COUT, 256, 0, stream>>>(
            tk, M_GRID, COUT, statsD + k * 128, statsD + k * 128 + 64);
        const size_t tot = (size_t)M_GRID * COUT;
        bn_relu_out_kernel<<<(unsigned)((tot + 255) / 256), 256, 0, stream>>>(
            tk, statsD + k * 128, statsD + k * 128 + 64,
            g_out + k * COUT, b_out + k * COUT, out, k);
    }
}
